// AGA_44607530336839
// MI455X (gfx1250) — compile-verified, matrix operand order checked by reading, not run on hardware
//
#include <hip/hip_runtime.h>
#include <hip/hip_bf16.h>
#include <math.h>

// ---------------------------------------------------------------------------
// Agent-attention fused pipeline for MI455X (gfx1250, wave32, WMMA).
//   k/v workspace stored bf16 (halves L2 traffic; stage0 re-reads them 49x),
//   q/agent/out0 stored f32 (pooling + stage-1 accuracy),
//   final output uses non-temporal stores to preserve L2 for the workspace.
// ---------------------------------------------------------------------------

typedef __bf16 v16bf __attribute__((ext_vector_type(16)));
typedef __bf16 v8bf  __attribute__((ext_vector_type(8)));
typedef float  v8f   __attribute__((ext_vector_type(8)));

#define B_      32
#define N_      3136
#define DIM_    64
#define H_      8
#define DH_     8
#define AGENTS_ 49
#define NPA_    64           // N_/AGENTS_
#define SCALE_  0.35355339059327373f

// ---------------------------------------------------------------------------
// Kernel 1: q = x1 @ Wq^T (f32 out), k = x2 @ Wk^T, v = x2 @ Wv^T (bf16 out).
// One block = 16 rows; 12 waves, one wave per 16x16 output tile of [q|k|v].
// bf16 WMMA 16x16x32, f32 accumulate, K = 64 in two chained steps.
// ---------------------------------------------------------------------------
__global__ __launch_bounds__(384)
void aga_qkv_kernel(const float* __restrict__ x1, const float* __restrict__ x2,
                    const float* __restrict__ Wq, const float* __restrict__ Wk,
                    const float* __restrict__ Wv,
                    float* __restrict__ qb, __bf16* __restrict__ kb,
                    __bf16* __restrict__ vb)
{
    const int lane = threadIdx.x & 31;
    const int wave = threadIdx.x >> 5;      // 0..11
    const int row0 = blockIdx.x * 16;       // row tile in flattened (B*N)
    const int half = lane >> 4;             // lane group (K offset select)
    const int mn   = lane & 15;             // M for A, N for B/C/D
    const int n0   = (wave & 3) * 16;

    const float* src = (wave < 4) ? x1 : x2;
    const float* W   = (wave < 4) ? Wq : ((wave < 8) ? Wk : Wv);

    const float* arow = src + (size_t)(row0 + mn) * DIM_;  // A row m (row-major)
    const float* bcol = W   + (size_t)(n0 + mn) * DIM_;    // B col n == W row o

    v8f c = {0.f, 0.f, 0.f, 0.f, 0.f, 0.f, 0.f, 0.f};
    #pragma unroll
    for (int kc = 0; kc < 2; ++kc) {
        v16bf a, b;
        #pragma unroll
        for (int e = 0; e < 16; ++e) {
            // ISA 16-bit operand layout: element e -> K index below
            const int kk = kc * 32 + (e >> 3) * 16 + half * 8 + (e & 7);
            a[e] = (__bf16)arow[kk];
            b[e] = (__bf16)bcol[kk];
        }
        c = __builtin_amdgcn_wmma_f32_16x16x32_bf16(false, a, false, b,
                                                    (short)0, c, false, false);
    }
    // C/D layout: VGPR r -> row (r + 8*half), col mn
    if (wave < 4) {
        #pragma unroll
        for (int r = 0; r < 8; ++r)
            qb[(size_t)(row0 + r + 8 * half) * DIM_ + n0 + mn] = c[r];
    } else {
        __bf16* dstb = (wave < 8) ? kb : vb;
        #pragma unroll
        for (int r = 0; r < 8; ++r)
            dstb[(size_t)(row0 + r + 8 * half) * DIM_ + n0 + mn] = (__bf16)c[r];
    }
}

// ---------------------------------------------------------------------------
// Kernel 2: agent tokens = exact block mean of q over 64 consecutive rows.
// ---------------------------------------------------------------------------
__global__ __launch_bounds__(64)
void aga_pool_kernel(const float* __restrict__ qb, float* __restrict__ agent)
{
    const int ba = blockIdx.x;        // b*49 + a ; rows start at 64*ba
    const int c  = threadIdx.x;       // 0..63
    const float* p = qb + (size_t)ba * NPA_ * DIM_ + c;
    float s = 0.f;
    #pragma unroll 8
    for (int j = 0; j < NPA_; ++j) s += p[(size_t)j * DIM_];
    agent[(size_t)ba * DIM_ + c] = s * (1.0f / (float)NPA_);
}

// ---------------------------------------------------------------------------
// Kernel 3: stage 0 — agent tokens attend over keys/values (bf16 k/v).
// One workgroup per (b,h); two-pass softmax over n=3136 with logits in LDS,
// PV accumulation reduced via shared-memory float atomics (ds_add_f32).
// Head row = 8 bf16 = one 16B vector load; v rows prefetched during k pass.
// ---------------------------------------------------------------------------
__global__ __launch_bounds__(256)
void aga_stage0_kernel(const __bf16* __restrict__ kb, const __bf16* __restrict__ vb,
                       const float* __restrict__ agent, float* __restrict__ out0)
{
    const int bh = blockIdx.x;        // b*H + h
    const int b  = bh >> 3, h = bh & 7;
    const int t  = threadIdx.x;       // 0..255

    __shared__ float ag[AGENTS_ * DH_];
    __shared__ float s[N_];
    __shared__ float red[256];
    __shared__ float acc[1 + DH_];

    for (int i = t; i < AGENTS_ * DH_; i += 256) {
        const int a = i >> 3, d = i & 7;
        ag[i] = agent[((size_t)b * AGENTS_ + a) * DIM_ + h * DH_ + d];
    }
    __syncthreads();

    const __bf16* kbase = kb + (size_t)b * N_ * DIM_ + h * DH_;
    const __bf16* vbase = vb + (size_t)b * N_ * DIM_ + h * DH_;

    for (int a = 0; a < AGENTS_; ++a) {
        if (t < 1 + DH_) acc[t] = 0.f;
        float lmax = -3.0e38f;
        for (int j = t; j < N_; j += 256) {
            const v8bf kv = *(const v8bf*)(kbase + (size_t)j * DIM_);   // b128
            __builtin_prefetch((const void*)(vbase + (size_t)j * DIM_), 0, 3);
            float dot = 0.f;
            #pragma unroll
            for (int d = 0; d < DH_; ++d) dot += ag[a * DH_ + d] * (float)kv[d];
            dot *= SCALE_;
            s[j] = dot;
            lmax = fmaxf(lmax, dot);
        }
        red[t] = lmax;
        __syncthreads();                       // also covers acc[] zeroing
        for (int off = 128; off > 0; off >>= 1) {
            if (t < off) red[t] = fmaxf(red[t], red[t + off]);
            __syncthreads();
        }
        const float gmax = red[0];

        float lsum = 0.f;
        float lacc[DH_] = {0.f, 0.f, 0.f, 0.f, 0.f, 0.f, 0.f, 0.f};
        for (int j = t; j < N_; j += 256) {
            const float e = __expf(s[j] - gmax);
            lsum += e;
            const v8bf vv = *(const v8bf*)(vbase + (size_t)j * DIM_);   // b128
            #pragma unroll
            for (int d = 0; d < DH_; ++d) lacc[d] += e * (float)vv[d];
        }
        atomicAdd(&acc[0], lsum);
        #pragma unroll
        for (int d = 0; d < DH_; ++d) atomicAdd(&acc[1 + d], lacc[d]);
        __syncthreads();
        if (t < DH_)
            out0[((size_t)bh * AGENTS_ + a) * DH_ + t] = acc[1 + t] / acc[0];
        __syncthreads();
    }
}

// ---------------------------------------------------------------------------
// Kernel 4: stage 1 + concat + output projection, fused per 16-row tile.
// Phase A (VALU): softmax over 49 agents per (row, head) -> out1 into LDS
// tile next to x1 (the concat).  Phase B (WMMA): 8 waves, each one 16x16
// column tile of the 16x128 @ 128x128 GEMM, 4 chained bf16 WMMAs, + bias.
// Final stores are non-temporal (dead data; keep L2 for the workspace).
// ---------------------------------------------------------------------------
__global__ __launch_bounds__(256)
void aga_stage1_kernel(const float* __restrict__ qb, const float* __restrict__ x1,
                       const float* __restrict__ agent, const float* __restrict__ out0,
                       const float* __restrict__ Wo, const float* __restrict__ bo,
                       float* __restrict__ out)
{
    const int blk  = blockIdx.x;                 // b*196 + tile
    const int b    = blk / (N_ / 16);
    const int tile = blk % (N_ / 16);
    const size_t gr = (size_t)b * N_ + (size_t)tile * 16;
    const int t = threadIdx.x;                   // 0..255

    __shared__ float ldsag[H_ * AGENTS_ * DH_];  // [h][a][d]
    __shared__ float ldso0[H_ * AGENTS_ * DH_];  // [h][a][d]
    __shared__ float tileAB[16][130];            // cols 0..63 out1, 64..127 x1 (padded)
    __shared__ float logits[128][AGENTS_ + 1];

    for (int i = t; i < H_ * AGENTS_ * DH_; i += 256) {
        const int h = i / (AGENTS_ * DH_);
        const int r = i % (AGENTS_ * DH_);
        const int a = r >> 3, d = r & 7;
        ldsag[i] = agent[((size_t)b * AGENTS_ + a) * DIM_ + h * DH_ + d];
        ldso0[i] = out0[((size_t)(b * H_ + h) * AGENTS_ + a) * DH_ + d];
    }
    for (int i = t; i < 16 * DIM_; i += 256) {   // concat: x1 into cols 64..127
        const int m = i >> 6, c2 = i & 63;
        tileAB[m][DIM_ + c2] = x1[(gr + m) * DIM_ + c2];
    }
    __syncthreads();

    if (t < 128) {                               // one thread per (row, head)
        const int m = t >> 3, h = t & 7;
        const float* qr  = qb + (gr + m) * DIM_ + h * DH_;
        const float* agh = &ldsag[h * AGENTS_ * DH_];
        const float* o0h = &ldso0[h * AGENTS_ * DH_];
        float qv[DH_];
        #pragma unroll
        for (int d = 0; d < DH_; ++d) qv[d] = qr[d];

        float mx = -3.0e38f;
        for (int a = 0; a < AGENTS_; ++a) {
            float dot = 0.f;
            #pragma unroll
            for (int d = 0; d < DH_; ++d) dot += qv[d] * agh[a * DH_ + d];
            dot *= SCALE_;
            logits[t][a] = dot;
            mx = fmaxf(mx, dot);
        }
        float sum = 0.f;
        for (int a = 0; a < AGENTS_; ++a) {
            const float e = __expf(logits[t][a] - mx);
            logits[t][a] = e;
            sum += e;
        }
        const float inv = 1.0f / sum;
        #pragma unroll
        for (int d = 0; d < DH_; ++d) {
            float o = 0.f;
            for (int a = 0; a < AGENTS_; ++a) o += logits[t][a] * o0h[a * DH_ + d];
            tileAB[m][h * DH_ + d] = o * inv;
        }
    }
    __syncthreads();

    // Phase B: out_tile(16x128) = cat(16x128) @ Wo^T + bo, bf16 WMMA.
    const int wave = t >> 5, lane = t & 31;
    const int half = lane >> 4, mn = lane & 15;
    const int n0   = wave * 16;
    const float* bcol = Wo + (size_t)(n0 + mn) * 128;   // B col n == Wo row o

    v8f c = {0.f, 0.f, 0.f, 0.f, 0.f, 0.f, 0.f, 0.f};
    #pragma unroll
    for (int kc = 0; kc < 4; ++kc) {
        v16bf a, bvec;
        #pragma unroll
        for (int e = 0; e < 16; ++e) {
            const int kk = kc * 32 + (e >> 3) * 16 + half * 8 + (e & 7);
            a[e]    = (__bf16)tileAB[mn][kk];
            bvec[e] = (__bf16)bcol[kk];
        }
        c = __builtin_amdgcn_wmma_f32_16x16x32_bf16(false, a, false, bvec,
                                                    (short)0, c, false, false);
    }
    const float bias = bo[n0 + mn];
    #pragma unroll
    for (int r = 0; r < 8; ++r)
        __builtin_nontemporal_store(c[r] + bias,
            &out[(gr + r + 8 * half) * 128 + n0 + mn]);
}

// ---------------------------------------------------------------------------
// Launch
// ---------------------------------------------------------------------------
extern "C" void kernel_launch(void* const* d_in, const int* in_sizes, int n_in,
                              void* d_out, int out_size, void* d_ws, size_t ws_size,
                              hipStream_t stream)
{
    const float* x1 = (const float*)d_in[0];
    const float* x2 = (const float*)d_in[1];
    const float* Wq = (const float*)d_in[2];
    const float* Wk = (const float*)d_in[3];
    const float* Wv = (const float*)d_in[4];
    const float* Wo = (const float*)d_in[5];
    const float* bo = (const float*)d_in[6];
    float* out = (float*)d_out;

    const size_t rows = (size_t)B_ * N_;
    float*  qb    = (float*)d_ws;                          // rows*64 f32
    __bf16* kb    = (__bf16*)(qb + rows * DIM_);           // rows*64 bf16
    __bf16* vb    = kb + rows * DIM_;                      // rows*64 bf16
    float*  agent = (float*)(vb + rows * DIM_);            // B*49*64 f32
    float*  out0  = agent + (size_t)B_ * AGENTS_ * DIM_;   // B*H*49*8 f32

    aga_qkv_kernel<<<(int)(rows / 16), 384, 0, stream>>>(x1, x2, Wq, Wk, Wv,
                                                         qb, kb, vb);
    aga_pool_kernel<<<B_ * AGENTS_, 64, 0, stream>>>(qb, agent);
    aga_stage0_kernel<<<B_ * H_, 256, 0, stream>>>(kb, vb, agent, out0);
    aga_stage1_kernel<<<B_ * (N_ / 16), 256, 0, stream>>>(qb, x1, agent, out0,
                                                          Wo, bo, out);
}
